// TrajectoryDecoder_120259085201
// MI455X (gfx1250) — compile-verified, matrix operand order checked by reading, not run on hardware
//
#include <hip/hip_runtime.h>
#include <math.h>

// ---- problem constants ----
#define BB 16
#define DD 64
#define TT 256
#define HH 128
#define H2c 64
#define BDT (BB*DD*TT)          // 262144

constexpr int FLD = 132;  // fp32 fused buffer ld (pad: stride%64 dwords != 0)
constexpr int HLD = 68;   // fp32 head buffer ld

typedef __attribute__((ext_vector_type(16))) __bf16 v16bf;
typedef __attribute__((ext_vector_type(8)))  __bf16 v8bf;
typedef __attribute__((ext_vector_type(8)))  float  v8f;

// ============ packed fragment layout ============
// One 16x16(xK-step-32) fragment tile = 32 lanes x 16 bf16, stored as:
//   [tile*512 + 0   .. 255] : per-lane lo-octet  (frag elems t=0..7 ), lane stride 8 (16B)
//   [tile*512 + 256 .. 511] : per-lane hi-octet  (frag elems t=8..15)
// Semantic k for elem t of lane (half=lane>>4): k = ((t&8)<<1) + (t&7) + half*8.
// => octet (thi, half) covers k in [thi*16 + half*8, +8).

__device__ inline v16bf load_frag(const __bf16* P, int tile, int lane) {
  v8bf lo = *(const v8bf*)(P + tile * 512 + lane * 8);
  v8bf hi = *(const v8bf*)(P + tile * 512 + 256 + lane * 8);
  return __builtin_shufflevector(lo, hi, 0,1,2,3,4,5,6,7,8,9,10,11,12,13,14,15);
}

__device__ inline void store_oct(__bf16* P, int tile, int thi, int lane, v8bf v) {
  *(v8bf*)(P + tile * 512 + thi * 256 + lane * 8) = v;
}
__device__ inline v8bf cvt8(const float* s) {
  v8bf v;
#pragma unroll
  for (int i = 0; i < 8; ++i) v[i] = (__bf16)s[i];
  return v;
}

// pack A (rows x 128, f32 row-major ld) into frag-major; tiles indexed (tm*KS + ks)
template <int KS>
__device__ inline void packA_f32(__bf16* Ap, const float* src, int ldsrc, int rows) {
  for (int o = threadIdx.x; o < rows * 16; o += 256) {
    int m = o >> 4, oc = o & 15;                       // oc indexes k-octet (k8 = oc*8)
    int ks = oc >> 2, half = oc & 1, thi = (oc >> 1) & 1;
    int lane = (m & 15) + (half << 4);
    store_oct(Ap, (m >> 4) * KS + ks, thi, lane, cvt8(src + (size_t)m * ldsrc + oc * 8));
  }
}
// pack B from *transposed* f32 weight: B[k][n] = w[n*ldw + k]; tiles (ks*NT + tn)
template <int NT>
__device__ inline void packBT_f32(__bf16* Bp, const float* w, int ldw, int ncols) {
  for (int o = threadIdx.x; o < ncols * 16; o += 256) {
    int n = o >> 4, oc = o & 15;
    int ks = oc >> 2, half = oc & 1, thi = (oc >> 1) & 1;
    int lane = (n & 15) + (half << 4);
    store_oct(Bp, ks * NT + (n >> 4), thi, lane, cvt8(w + (size_t)n * ldw + oc * 8));
  }
}

// ============ generic packed block GEMM ============
// 256 thr = 8 waves; wave w owns tile-rows {w, w+8,...}; epilogue runs after an
// internal barrier so it may overwrite operand buffers.
template <int MREP, int NT, int KS, typename F>
__device__ inline void gemm_block(const __bf16* Ap, const __bf16* Bp, F&& epi) {
  const int lane = threadIdx.x & 31, wave = threadIdx.x >> 5;
  v8f acc[MREP][NT];
  v8f zero = {};
#pragma unroll
  for (int rp = 0; rp < MREP; ++rp)
#pragma unroll
    for (int tn = 0; tn < NT; ++tn) acc[rp][tn] = zero;

  for (int ks = 0; ks < KS; ++ks) {
    v16bf bfrag[NT];
#pragma unroll
    for (int tn = 0; tn < NT; ++tn) bfrag[tn] = load_frag(Bp, ks * NT + tn, lane);
#pragma unroll
    for (int rp = 0; rp < MREP; ++rp) {
      v16bf afrag = load_frag(Ap, (wave + rp * 8) * KS + ks, lane);
#pragma unroll
      for (int tn = 0; tn < NT; ++tn)
        acc[rp][tn] = __builtin_amdgcn_wmma_f32_16x16x32_bf16(
            false, afrag, false, bfrag[tn], (short)0, acc[rp][tn], false, false);
    }
  }
  __syncthreads();
  const int half = (lane >> 4) & 1, nlo = lane & 15;
#pragma unroll
  for (int rp = 0; rp < MREP; ++rp) {
    const int tm = wave + rp * 8;
#pragma unroll
    for (int tn = 0; tn < NT; ++tn)
#pragma unroll
      for (int r = 0; r < 8; ++r)
        epi(tm * 16 + r + 8 * half, tn * 16 + nlo, acc[rp][tn][r]);
  }
  __syncthreads();
}

__device__ inline float gelu_exact(float x) {
  return 0.5f * x * (1.0f + erff(x * 0.70710678118654752f));
}

__device__ inline void layernorm_rows(float* sm, const float* g, const float* be) {
  float* row = sm + (size_t)threadIdx.x * FLD;      // 256 threads == 256 rows
  float m = 0.f;
  for (int h = 0; h < HH; h += 4) {
    float4 v = *(const float4*)(row + h);
    m += v.x + v.y + v.z + v.w;
  }
  m *= (1.f / HH);
  float var = 0.f;
  for (int h = 0; h < HH; h += 4) {
    float4 v = *(const float4*)(row + h);
    float a = v.x - m, b = v.y - m, c = v.z - m, d = v.w - m;
    var += a * a + b * b + c * c + d * d;
  }
  var *= (1.f / HH);
  const float rs = rsqrtf(var + 1e-5f);
  for (int h = 0; h < HH; h += 4) {
    float4 v = *(const float4*)(row + h);
    float4 gg = *(const float4*)(g + h);
    float4 bb = *(const float4*)(be + h);
    v.x = (v.x - m) * rs * gg.x + bb.x;
    v.y = (v.y - m) * rs * gg.y + bb.y;
    v.z = (v.z - m) * rs * gg.z + bb.z;
    v.w = (v.w - m) * rs * gg.w + bb.w;
    *(float4*)(row + h) = v;
  }
  __syncthreads();
}

// ---------- kernel 0: branch linear bias rows: br_ws[bd,h]=branch[bd]·W_br[h]+b_br[h] ----------
__global__ void bias_kernel(const float* __restrict__ branch, const float* __restrict__ W_br,
                            const float* __restrict__ b_br, float* __restrict__ br_ws) {
  __shared__ float xs[HH];
  const int row = blockIdx.x, h = threadIdx.x;
  xs[h] = branch[(size_t)row * HH + h];
  __syncthreads();
  float a = b_br[h];
  const float* wr = W_br + (size_t)h * HH;
  for (int j = 0; j < HH; j += 4) {
    float4 x = *(const float4*)(xs + j);
    float4 w = *(const float4*)(wr + j);
    a += x.x * w.x + x.y * w.y + x.z * w.z + x.w * w.w;
  }
  br_ws[(size_t)row * HH + h] = a;
}

// ---------- kernel 1: tmp[bd,h,j] = sum_i branch[bd,i]*W_bil[h,i,j] (bf16) ----------
__global__ void tmp_kernel(const float* __restrict__ branch, const float* __restrict__ W_bil,
                           void* tmp_ws) {
  extern __shared__ char smem[];
  __bf16* ap = (__bf16*)smem;                       // 8*4 tiles  = 32768 B
  __bf16* bp = (__bf16*)(smem + 32768);             // 4*8 tiles  = 32768 B
  const int bt = blockIdx.x, h = blockIdx.y, tid = threadIdx.x;

  packA_f32<4>(ap, branch + (size_t)bt * 128 * HH, HH, 128);
  // B[k=i][n=j] = W_bil[h,i,j]: k-strided source (L1-resident slice)
  const float* Wb = W_bil + (size_t)h * HH * HH;
  for (int o = tid; o < HH * 16; o += 256) {
    int n = o >> 4, oc = o & 15;
    int ks = oc >> 2, half = oc & 1, thi = (oc >> 1) & 1;
    int lane = (n & 15) + (half << 4);
    const float* s = Wb + (size_t)(oc * 8) * HH + n;
    v8bf v;
#pragma unroll
    for (int i = 0; i < 8; ++i) v[i] = (__bf16)s[(size_t)i * HH];
    store_oct(bp, ks * 8 + (n >> 4), thi, lane, v);
  }
  __syncthreads();

  __bf16* outp = (__bf16*)tmp_ws;
  gemm_block<1, 8, 4>(ap, bp, [&](int row, int col, float v) {
    outp[((size_t)bt * 128 + row) * (HH * HH) + (size_t)h * HH + col] = (__bf16)v;
  });
}

// ---------- kernel 2: fully fused decoder, one workgroup per (b,d) ----------
__global__ void main_kernel(const float* __restrict__ trunk, const void* tmp_ws,
                            const float* __restrict__ br_ws,
                            const float* __restrict__ b_bil, const float* __restrict__ W_tr,
                            const float* __restrict__ b_tr,
                            const float* __restrict__ g_fn, const float* __restrict__ beta_fn,
                            const float* __restrict__ res_w1, const float* __restrict__ res_b1,
                            const float* __restrict__ res_w2, const float* __restrict__ res_b2,
                            const float* __restrict__ res_g, const float* __restrict__ res_beta,
                            const float* __restrict__ pos_w1, const float* __restrict__ pos_b1,
                            const float* __restrict__ pos_w2, const float* __restrict__ pos_b2,
                            const float* __restrict__ ex_w1, const float* __restrict__ ex_b1,
                            const float* __restrict__ ex_w2, const float* __restrict__ ex_b2,
                            const float* __restrict__ mask, float* __restrict__ out) {
  const int bd = blockIdx.x, b = bd >> 6, tid = threadIdx.x;
  extern __shared__ char smem[];
  float*  sm_fused = (float*)smem;                               // 256 x FLD fp32 (135168)
  __bf16* sm_ap = (__bf16*)(smem + 256 * FLD * 4);               // 16*4 tiles (65536)
  __bf16* sm_bp = (__bf16*)((char*)sm_ap + 65536);               // 4*8 tiles  (32768)
  float*  sm_hd = (float*)((char*)sm_bp + 32768);                // 256 x HLD fp32 (69632)
  float*  sm_cb = (float*)((char*)sm_hd + 256 * HLD * 4);        // 128 fp32 (512)

  const __bf16* tmp_bd = (const __bf16*)tmp_ws + (size_t)bd * (HH * HH);

  // ---- stage fused0 operands: A = trunk[b] (256x128); B[j][h] = tmp_bd[h,j] + W_tr[h,j]
  packA_f32<4>(sm_ap, trunk + (size_t)b * TT * HH, HH, TT);
  for (int o = tid; o < HH * 16; o += 256) {          // B pack: n=h, k=j (both contiguous)
    int n = o >> 4, oc = o & 15;
    int ks = oc >> 2, half = oc & 1, thi = (oc >> 1) & 1;
    int lane = (n & 15) + (half << 4);
    const __bf16* tp = tmp_bd + (size_t)n * HH + oc * 8;
    const float*  wp = W_tr + (size_t)n * HH + oc * 8;
    v8bf v;
#pragma unroll
    for (int i = 0; i < 8; ++i) v[i] = (__bf16)((float)tp[i] + wp[i]);
    store_oct(sm_bp, ks * 8 + (n >> 4), thi, lane, v);
  }
  if (tid < HH) sm_cb[tid] = b_bil[tid] + b_tr[tid] + br_ws[(size_t)bd * HH + tid];
  __syncthreads();

  gemm_block<2, 8, 4>(sm_ap, sm_bp, [&](int row, int col, float v) {
    sm_fused[row * FLD + col] = v + sm_cb[col];
  });
  layernorm_rows(sm_fused, g_fn, beta_fn);

  // ---- residual MLP blocks ----
  for (int i = 0; i < 2; ++i) {
    packA_f32<4>(sm_ap, sm_fused, FLD, TT);
    packBT_f32<8>(sm_bp, res_w1 + (size_t)i * HH * HH, HH, HH);
    if (tid < HH) sm_cb[tid] = res_b1[i * HH + tid];
    __syncthreads();
    gemm_block<2, 8, 4>(sm_ap, sm_bp, [&](int row, int col, float v) {
      // overwrite A-operand in place (packed layout) -- safe: internal barrier
      float gv = gelu_exact(v + sm_cb[col]);
      int oc = col >> 3, ks = oc >> 2, half = oc & 1, thi = (oc >> 1) & 1;
      int lane = (row & 15) + (half << 4);
      sm_ap[(((row >> 4) * 4 + ks) * 512) + thi * 256 + lane * 8 + (col & 7)] = (__bf16)gv;
    });
    packBT_f32<8>(sm_bp, res_w2 + (size_t)i * HH * HH, HH, HH);
    if (tid < HH) sm_cb[tid] = res_b2[i * HH + tid];
    __syncthreads();
    gemm_block<2, 8, 4>(sm_ap, sm_bp, [&](int row, int col, float v) {
      sm_fused[row * FLD + col] += v + sm_cb[col];     // residual add
    });
    layernorm_rows(sm_fused, res_g + i * HH, res_beta + i * HH);
  }

  // ---- heads ----
  packA_f32<4>(sm_ap, sm_fused, FLD, TT);
  __syncthreads();
  const float mval = mask[bd];

  for (int head = 0; head < 2; ++head) {
    const float* w1  = head ? ex_w1 : pos_w1;
    const float* bb1 = head ? ex_b1 : pos_b1;
    const float* w2  = head ? ex_w2 : pos_w2;
    const float* bb2 = head ? ex_b2 : pos_b2;
    packBT_f32<4>(sm_bp, w1, HH, H2c);
    if (tid < H2c) sm_cb[tid] = bb1[tid];
    __syncthreads();
    gemm_block<2, 4, 4>(sm_ap, sm_bp, [&](int row, int col, float v) {
      sm_hd[row * HLD + col] = gelu_exact(v + sm_cb[col]);
    });
    // final 64->1 projection, one row per thread
    float s = bb2[0];
    const float* hr = sm_hd + (size_t)tid * HLD;
    for (int c = 0; c < H2c; c += 4) {
      float4 hv = *(const float4*)(hr + c);
      float4 wv = *(const float4*)(w2 + c);
      s += hv.x * wv.x + hv.y * wv.y + hv.z * wv.z + hv.w * wv.w;
    }
    float o = head ? (1.f / (1.f + __expf(-s))) : fminf(fmaxf(s, 0.f), 1.f);
    out[(size_t)head * BDT + (size_t)bd * TT + tid] = o * mval;
    __syncthreads();
  }
}

extern "C" void kernel_launch(void* const* d_in, const int* in_sizes, int n_in,
                              void* d_out, int out_size, void* d_ws, size_t ws_size,
                              hipStream_t stream) {
  (void)in_sizes; (void)n_in; (void)out_size; (void)ws_size;
  const float* branch  = (const float*)d_in[0];
  const float* trunk   = (const float*)d_in[1];
  const float* mask    = (const float*)d_in[2];
  const float* W_bil   = (const float*)d_in[3];
  const float* b_bil   = (const float*)d_in[4];
  const float* W_br    = (const float*)d_in[5];
  const float* b_br    = (const float*)d_in[6];
  const float* W_tr    = (const float*)d_in[7];
  const float* b_tr    = (const float*)d_in[8];
  const float* g_fn    = (const float*)d_in[9];
  const float* beta_fn = (const float*)d_in[10];
  const float* res_w1  = (const float*)d_in[11];
  const float* res_b1  = (const float*)d_in[12];
  const float* res_w2  = (const float*)d_in[13];
  const float* res_b2  = (const float*)d_in[14];
  const float* res_g   = (const float*)d_in[15];
  const float* res_beta= (const float*)d_in[16];
  const float* pos_w1  = (const float*)d_in[17];
  const float* pos_b1  = (const float*)d_in[18];
  const float* pos_w2  = (const float*)d_in[19];
  const float* pos_b2  = (const float*)d_in[20];
  const float* ex_w1   = (const float*)d_in[21];
  const float* ex_b1   = (const float*)d_in[22];
  const float* ex_w2   = (const float*)d_in[23];
  const float* ex_b2   = (const float*)d_in[24];

  char* ws = (char*)d_ws;
  const size_t tmp_bytes = (size_t)BB * DD * HH * HH * 2;   // 32 MB bf16
  void*  ws_tmp = (void*)ws;
  float* ws_br  = (float*)(ws + tmp_bytes);                  // 1024*128 fp32

  bias_kernel<<<BB * DD, HH, 0, stream>>>(branch, W_br, b_br, ws_br);

  dim3 g1(8, HH);
  tmp_kernel<<<g1, 256, 65536, stream>>>(branch, W_bil, ws_tmp);

  const size_t lds2 = (size_t)256 * FLD * 4 + 65536 + 32768 +
                      (size_t)256 * HLD * 4 + 512;          // 303616 B
  main_kernel<<<BB * DD, 256, lds2, stream>>>(trunk, ws_tmp, ws_br, b_bil, W_tr, b_tr,
                                              g_fn, beta_fn, res_w1, res_b1, res_w2, res_b2,
                                              res_g, res_beta, pos_w1, pos_b1, pos_w2, pos_b2,
                                              ex_w1, ex_b1, ex_w2, ex_b2, mask,
                                              (float*)d_out);
}
